// DynamicTemporalAttention_42949672961999
// MI455X (gfx1250) — matrix operand order repaired and checked by reading, not compiled, not run on hardware
//
#include <hip/hip_runtime.h>

// ---------------------------------------------------------------------------
// MI455X (gfx1250) fused attention: bf16 WMMA (16x16x32, f32 accum) everywhere,
// tile staging via gfx1250 async-to-LDS (GEMMs) and Tensor Data Mover (flash).
// B=2, S=2048, D=1024, H=16, Dh=64.
// ---------------------------------------------------------------------------

typedef __attribute__((ext_vector_type(16))) __bf16 v16bf;
typedef __attribute__((ext_vector_type(8)))  __bf16 v8bf;
typedef __attribute__((ext_vector_type(4)))  __bf16 v4bf;
typedef __attribute__((ext_vector_type(8)))  float  v8f;
typedef __attribute__((ext_vector_type(4)))  unsigned int u32x4;
typedef __attribute__((ext_vector_type(8)))  unsigned int u32x8;

#define DEV __device__ __forceinline__

DEV v8f wmma_bf16(v16bf a, v16bf b, v8f c) {
  // D = A(16x32) * B(32x16) + C, f32 accumulate
  return __builtin_amdgcn_wmma_f32_16x16x32_bf16(false, a, false, b,
                                                 (short)0, c, false, false);
}

// Per-lane async global->LDS 16B copy (ASYNCcnt). lds_off = LDS byte address.
DEV void async_b128(unsigned int lds_off, const __bf16* g) {
  asm volatile("global_load_async_to_lds_b128 %0, %1, off"
               :: "v"(lds_off), "v"(g)
               : "memory");
}
DEV void wait_asynccnt0() { asm volatile("s_wait_asynccnt 0x0" ::: "memory"); }

// A-fragment (16x32, bf16), tile is row-major [row][k] with leading dim ld.
// lanes 0-15: row=lane, elems 0..7 = K 0..7, elems 8..15 = K 16..23
// lanes 16-31: row=lane-16, elems 0..7 = K 8..15, elems 8..15 = K 24..31
DEV v16bf frag_a(const __bf16* tile, int ld, int row0, int k0) {
  const int lane = (int)threadIdx.x & 31;
  const __bf16* p = tile + (row0 + (lane & 15)) * ld + k0 + ((lane >> 4) << 3);
  v8bf lo = *(const v8bf*)p;
  v8bf hi = *(const v8bf*)(p + 16);
  return __builtin_shufflevector(lo, hi, 0, 1, 2, 3, 4, 5, 6, 7,
                                 8, 9, 10, 11, 12, 13, 14, 15);
}

// B-fragment (32x16, bf16), tile is [n][k] row-major with leading dim ld.
// lanes 0-15: n=lane, elems 0..7 = K 0..7, 8..15 = K 16..23 ; lanes 16-31: n=lane-16, elems 0..7 = K 8..15, 8..15 = K 24..31
DEV v16bf frag_b(const __bf16* tile, int ld, int row0, int k0) {
  const int lane = (int)threadIdx.x & 31;
  const __bf16* p = tile + (row0 + (lane & 15)) * ld + k0 + ((lane >> 4) << 3);
  v8bf lo = *(const v8bf*)p;
  v8bf hi = *(const v8bf*)(p + 16);
  return __builtin_shufflevector(lo, hi, 0, 1, 2, 3, 4, 5, 6, 7,
                                 8, 9, 10, 11, 12, 13, 14, 15);
}

// ---------------------------------------------------------------------------
// fp32 -> bf16 flat convert (exactly 4 elems/thread)
// ---------------------------------------------------------------------------
__global__ __launch_bounds__(256) void k_cvt(const float* __restrict__ in,
                                             __bf16* __restrict__ outp) {
  const size_t i = ((size_t)blockIdx.x * 256 + threadIdx.x) * 4;
  const float4 f = *(const float4*)(in + i);
  v4bf o4 = {(__bf16)f.x, (__bf16)f.y, (__bf16)f.z, (__bf16)f.w};
  *(v4bf*)(outp + i) = o4;
}

// ---------------------------------------------------------------------------
// W[k][n] fp32 (1024x1024) -> Wt[n][k] bf16 (transpose+convert), 32x32 tiles
// ---------------------------------------------------------------------------
__global__ __launch_bounds__(256) void k_transpose(const float* __restrict__ W,
                                                   __bf16* __restrict__ Wt) {
  __shared__ float t[32][33];
  const int bx = (int)blockIdx.x * 32;  // n base
  const int by = (int)blockIdx.y * 32;  // k base
  const int tx = (int)threadIdx.x & 31;
  const int ty = (int)threadIdx.x >> 5;
#pragma unroll
  for (int ii = 0; ii < 4; ++ii) {
    const int i = ty + ii * 8;
    t[i][tx] = W[(size_t)(by + i) * 1024 + bx + tx];
  }
  __syncthreads();
#pragma unroll
  for (int ii = 0; ii < 4; ++ii) {
    const int i = ty + ii * 8;
    Wt[(size_t)(bx + i) * 1024 + by + tx] = (__bf16)t[tx][i];
  }
}

// ---------------------------------------------------------------------------
// Fused QKV projection: A = xb [4096 x 1024] bf16, Bt = WqkvT [3072 x 1024]
// bf16 (Bt row n holds column n of W). 128x128 block tile, 256 thr = 8 waves,
// each wave computes 64x32 via 4x2 WMMA tiles. LDS tiles filled with
// global_load_async_to_lds_b128 (ASYNCcnt). Scatter bf16 to [B,H,S,Dh].
// ---------------------------------------------------------------------------
__global__ __launch_bounds__(256) void k_gemm_qkv(
    const __bf16* __restrict__ A, const __bf16* __restrict__ Bt,
    const float* __restrict__ bq, const float* __restrict__ bk,
    const float* __restrict__ bv, __bf16* __restrict__ qd,
    __bf16* __restrict__ kd, __bf16* __restrict__ vd) {
  constexpr int KD = 1024, LDT = 40;
  alignas(16) __shared__ __bf16 As[128 * LDT];
  alignas(16) __shared__ __bf16 Bs[128 * LDT];
  const int tid = (int)threadIdx.x;
  const int w = tid >> 5, lane = tid & 31;
  const int wm = (w >> 2) * 64, wn = (w & 3) * 32;
  const int Mblk = (int)blockIdx.y * 128, Nblk = (int)blockIdx.x * 128;
  const int lr = tid >> 1, ls = (tid & 1) << 4;
  const v8f zero = {0, 0, 0, 0, 0, 0, 0, 0};
  v8f acc[4][2];
#pragma unroll
  for (int mt = 0; mt < 4; ++mt)
#pragma unroll
    for (int nt = 0; nt < 2; ++nt) acc[mt][nt] = zero;

  const unsigned int asOff = (unsigned int)(size_t)(void*)&As[lr * LDT + ls];
  const unsigned int bsOff = (unsigned int)(size_t)(void*)&Bs[lr * LDT + ls];
  const __bf16* ga = A + (size_t)(Mblk + lr) * KD + ls;
  const __bf16* gb = Bt + (size_t)(Nblk + lr) * KD + ls;
  for (int k0 = 0; k0 < KD; k0 += 32) {
    __syncthreads();  // WAR: previous tile fully consumed by all waves
    async_b128(asOff, ga);
    async_b128(asOff + 16, ga + 8);
    async_b128(bsOff, gb);
    async_b128(bsOff + 16, gb + 8);
    if (k0 + 32 < KD) {  // gfx1250 global_prefetch_b8 for next k-tile
      __builtin_prefetch(ga + 32, 0, 1);
      __builtin_prefetch(gb + 32, 0, 1);
    }
    wait_asynccnt0();  // this wave's async copies landed in LDS
    __syncthreads();   // all waves' copies landed
    v16bf af[4], bfv[2];
#pragma unroll
    for (int mt = 0; mt < 4; ++mt) af[mt] = frag_a(As, LDT, wm + mt * 16, 0);
#pragma unroll
    for (int nt = 0; nt < 2; ++nt) bfv[nt] = frag_b(Bs, LDT, wn + nt * 16, 0);
#pragma unroll
    for (int mt = 0; mt < 4; ++mt)
#pragma unroll
      for (int nt = 0; nt < 2; ++nt)
        acc[mt][nt] = wmma_bf16(af[mt], bfv[nt], acc[mt][nt]);
    ga += 32;
    gb += 32;
  }
  // Epilogue: m = b*2048+s ; n in [0,3072): which=n>>10, h=(n&1023)>>6, dh=n&63
#pragma unroll
  for (int mt = 0; mt < 4; ++mt)
#pragma unroll
    for (int nt = 0; nt < 2; ++nt)
#pragma unroll
      for (int e = 0; e < 8; ++e) {
        const int m = Mblk + wm + mt * 16 + e + ((lane >> 4) << 3);
        const int n = Nblk + wn + nt * 16 + (lane & 15);
        const int bb = m >> 11, s = m & 2047;
        const int which = n >> 10, nn = n & 1023;
        const int h = nn >> 6, dh = nn & 63;
        const float* bp = (which == 0) ? bq : (which == 1) ? bk : bv;
        __bf16* dst = (which == 0) ? qd : (which == 1) ? kd : vd;
        const float val = acc[mt][nt][e] + bp[nn];
        dst[(((size_t)bb * 16 + h) * 2048 + s) * 64 + dh] = (__bf16)val;
      }
}

// ---------------------------------------------------------------------------
// Flash attention with ALiBi-style bias. One block = (b,h, 128 Q rows).
// 8 waves, each owns 16 Q rows; K streamed via TDM tensor_load_to_lds with
// the pad engine reproducing the padded LDS pitch (72 elems = 144 B rows);
// V transposed through VGPRs. Online softmax in f32; P re-laid out bf16 in
// per-wave LDS for the PV WMMA.
// ---------------------------------------------------------------------------
__global__ __launch_bounds__(256) void k_flash(
    const __bf16* __restrict__ q, const __bf16* __restrict__ kbuf,
    const __bf16* __restrict__ vbuf, const float* __restrict__ slopes,
    __bf16* __restrict__ ob) {
  constexpr int LDK = 72;
  alignas(16) __shared__ __bf16 Kt[64 * LDK];      // [t][dh]   (TDM-filled)
  alignas(16) __shared__ __bf16 Vt[64 * LDK];      // [dh][t]   (V transposed)
  alignas(16) __shared__ __bf16 Pw[8 * 16 * LDK];  // per-wave P [16][t]
  const int tid = (int)threadIdx.x;
  const int w = tid >> 5, lane = tid & 31;
  const int hi8 = (lane >> 4) << 3;
  const int bid = (int)blockIdx.x;
  const int qt = bid & 15;  // S/128 = 16 q-tiles
  const int bh = bid >> 4;  // 0..31
  const int h = bh & 15;
  const size_t base = (size_t)bh * (2048 * 64);
  const int q0 = qt * 128;
  const float coef = -logf(1.0f + __expf(slopes[h]));  // -softplus(slope)
  const float scale = 0.125f;                          // 1/sqrt(64)

  // ---- TDM descriptor (D#) for one 64x64 bf16 K tile -> padded LDS ----
  // group0: count=1 | lds_addr | global_addr[56:0] | type=2
  // group1: data_size=2B, pad_enable, pad_interval=32 DW (code 4),
  //         pad_amount=4 DW (code 3) -> LDS row pitch 144 B = LDK elems;
  //         tensor_dim0=64, tensor_dim1=2048, tile 64x64, dim0_stride=64.
  const unsigned long long gK = (unsigned long long)(size_t)(kbuf + base);
  u32x4 g0;
  g0[0] = 1u;  // count=1, user descriptor
  g0[1] = (unsigned int)(size_t)(void*)Kt;
  g0[2] = 0u;
  g0[3] = 0u;
  u32x8 g1;
  g1[0] = (1u << 16) | (1u << 20) | (4u << 22) | (3u << 25);
  g1[1] = (64u << 16);    // tensor_dim0[15:0]=64
  g1[2] = (2048u << 16);  // tensor_dim0[31:16]=0 | tensor_dim1[15:0]=2048
  g1[3] = (64u << 16);    // tensor_dim1[31:16]=0 | tile_dim0=64
  g1[4] = 64u;            // tile_dim1=64, tile_dim2=0
  g1[5] = 64u;            // tensor_dim0_stride[31:0]=64
  g1[6] = 0u;
  g1[7] = 0u;

  // Q fragments for this wave's 16 rows, held in registers all kernel long.
  const __bf16* qtile = q + base + (size_t)(q0 + w * 16) * 64;
  const v16bf qf0 = frag_a(qtile, 64, 0, 0);
  const v16bf qf1 = frag_a(qtile, 64, 0, 32);

  const v8f vzero = {0, 0, 0, 0, 0, 0, 0, 0};
  float mrow[8], lrow[8];
  v8f o[4];
#pragma unroll
  for (int e = 0; e < 8; ++e) {
    mrow[e] = -1e30f;
    lrow[e] = 0.0f;
  }
#pragma unroll
  for (int d = 0; d < 4; ++d) o[d] = vzero;

  const int r = tid >> 2, sseg = (tid & 3) << 4;  // V tile loader mapping
  __bf16* pw = &Pw[w * 16 * LDK];

  for (int t0 = 0; t0 < 2048; t0 += 64) {
    // ---- V tile into registers (pre-barrier) ----
    const __bf16* gv = vbuf + base + (size_t)(t0 + r) * 64 + sseg;
    v8bf va = *(const v8bf*)gv, vc = *(const v8bf*)(gv + 8);
    // ---- point D# at this K tile (all scalar math) ----
    const unsigned long long ga = gK + (unsigned long long)t0 * 128u;
    g0[2] = (unsigned int)ga;
    g0[3] = (unsigned int)((ga >> 32) & 0x01FFFFFFull) | (2u << 30);
    __syncthreads();  // WAR: previous Kt/Vt fully consumed
    if (w == 0)       // wave-uniform branch: one TDM op per workgroup
      asm volatile("tensor_load_to_lds %0, %1" :: "s"(g0), "s"(g1) : "memory");
#pragma unroll
    for (int i = 0; i < 8; ++i) {  // V^T scatter into LDS
      Vt[(sseg + i) * LDK + r] = va[i];
      Vt[(sseg + 8 + i) * LDK + r] = vc[i];
    }
    __builtin_amdgcn_s_wait_tensorcnt(0);  // K tile landed (issuer wave)
    __syncthreads();                       // visible to all waves

    // ---- S = (Q K^T) * scale + bias ----
    float sv[4][8];
#pragma unroll
    for (int nt = 0; nt < 4; ++nt) {
      v8f c = vzero;
      c = wmma_bf16(qf0, frag_b(Kt, LDK, nt * 16, 0), c);
      c = wmma_bf16(qf1, frag_b(Kt, LDK, nt * 16, 32), c);
      const int j = t0 + nt * 16 + (lane & 15);
#pragma unroll
      for (int e = 0; e < 8; ++e) {
        const int i = q0 + w * 16 + hi8 + e;
        sv[nt][e] = c[e] * scale + coef * fabsf((float)(i - j));
      }
    }

    // ---- online softmax: row = elem + 8*(lane>=16); cols across 16 lanes ----
#pragma unroll
    for (int e = 0; e < 8; ++e) {
      float rm = fmaxf(fmaxf(sv[0][e], sv[1][e]), fmaxf(sv[2][e], sv[3][e]));
#pragma unroll
      for (int off = 1; off <= 8; off <<= 1)
        rm = fmaxf(rm, __shfl_xor(rm, off, 32));  // stays in 16-lane half
      const float mn = fmaxf(mrow[e], rm);
      const float sc = __expf(mrow[e] - mn);
      float rs = 0.0f;
#pragma unroll
      for (int nt = 0; nt < 4; ++nt) {
        const float p = __expf(sv[nt][e] - mn);
        sv[nt][e] = p;
        rs += p;
      }
#pragma unroll
      for (int off = 1; off <= 8; off <<= 1) rs += __shfl_xor(rs, off, 32);
      lrow[e] = lrow[e] * sc + rs;
      mrow[e] = mn;
#pragma unroll
      for (int d = 0; d < 4; ++d) o[d][e] *= sc;
    }

    // ---- P (C-frag layout) -> per-wave LDS in A-frag row-major layout ----
#pragma unroll
    for (int nt = 0; nt < 4; ++nt)
#pragma unroll
      for (int e = 0; e < 8; ++e)
        pw[(hi8 + e) * LDK + nt * 16 + (lane & 15)] = (__bf16)sv[nt][e];
    // same-wave LDS ops are in-order: no barrier needed before re-reading pw

    // ---- O += P * V ----
    const v16bf pf0 = frag_a(pw, LDK, 0, 0);
    const v16bf pf1 = frag_a(pw, LDK, 0, 32);
#pragma unroll
    for (int d = 0; d < 4; ++d) {
      o[d] = wmma_bf16(pf0, frag_b(Vt, LDK, d * 16, 0), o[d]);
      o[d] = wmma_bf16(pf1, frag_b(Vt, LDK, d * 16, 32), o[d]);
    }
  }

  // ---- normalize and write bf16 O in [B,S,D] layout for the out-proj ----
  const int bb = bh >> 4;
#pragma unroll
  for (int d = 0; d < 4; ++d)
#pragma unroll
    for (int e = 0; e < 8; ++e) {
      const int s = q0 + w * 16 + hi8 + e;
      const int dd = h * 64 + d * 16 + (lane & 15);
      ob[((size_t)(bb * 2048 + s)) * 1024 + dd] = (__bf16)(o[d][e] / lrow[e]);
    }
}

// ---------------------------------------------------------------------------
// Output projection: obuf [4096x1024] bf16 x WoT [1024x1024] bf16 -> fp32 out
// (async-to-LDS tile staging, same structure as the QKV GEMM)
// ---------------------------------------------------------------------------
__global__ __launch_bounds__(256) void k_gemm_out(
    const __bf16* __restrict__ A, const __bf16* __restrict__ Bt,
    const float* __restrict__ bo, float* __restrict__ outp) {
  constexpr int KD = 1024, LDT = 40;
  alignas(16) __shared__ __bf16 As[128 * LDT];
  alignas(16) __shared__ __bf16 Bs[128 * LDT];
  const int tid = (int)threadIdx.x;
  const int w = tid >> 5, lane = tid & 31;
  const int wm = (w >> 2) * 64, wn = (w & 3) * 32;
  const int Mblk = (int)blockIdx.y * 128, Nblk = (int)blockIdx.x * 128;
  const int lr = tid >> 1, ls = (tid & 1) << 4;
  const v8f zero = {0, 0, 0, 0, 0, 0, 0, 0};
  v8f acc[4][2];
#pragma unroll
  for (int mt = 0; mt < 4; ++mt)
#pragma unroll
    for (int nt = 0; nt < 2; ++nt) acc[mt][nt] = zero;

  const unsigned int asOff = (unsigned int)(size_t)(void*)&As[lr * LDT + ls];
  const unsigned int bsOff = (unsigned int)(size_t)(void*)&Bs[lr * LDT + ls];
  const __bf16* ga = A + (size_t)(Mblk + lr) * KD + ls;
  const __bf16* gb = Bt + (size_t)(Nblk + lr) * KD + ls;
  for (int k0 = 0; k0 < KD; k0 += 32) {
    __syncthreads();
    async_b128(asOff, ga);
    async_b128(asOff + 16, ga + 8);
    async_b128(bsOff, gb);
    async_b128(bsOff + 16, gb + 8);
    if (k0 + 32 < KD) {
      __builtin_prefetch(ga + 32, 0, 1);
      __builtin_prefetch(gb + 32, 0, 1);
    }
    wait_asynccnt0();
    __syncthreads();
    v16bf af[4], bfv[2];
#pragma unroll
    for (int mt = 0; mt < 4; ++mt) af[mt] = frag_a(As, LDT, wm + mt * 16, 0);
#pragma unroll
    for (int nt = 0; nt < 2; ++nt) bfv[nt] = frag_b(Bs, LDT, wn + nt * 16, 0);
#pragma unroll
    for (int mt = 0; mt < 4; ++mt)
#pragma unroll
      for (int nt = 0; nt < 2; ++nt)
        acc[mt][nt] = wmma_bf16(af[mt], bfv[nt], acc[mt][nt]);
    ga += 32;
    gb += 32;
  }
#pragma unroll
  for (int mt = 0; mt < 4; ++mt)
#pragma unroll
    for (int nt = 0; nt < 2; ++nt)
#pragma unroll
      for (int e = 0; e < 8; ++e) {
        const int m = Mblk + wm + mt * 16 + e + ((lane >> 4) << 3);
        const int n = Nblk + wn + nt * 16 + (lane & 15);
        outp[(size_t)m * 1024 + n] = acc[mt][nt][e] + bo[n];
      }
}

// ---------------------------------------------------------------------------
extern "C" void kernel_launch(void* const* d_in, const int* in_sizes, int n_in,
                              void* d_out, int out_size, void* d_ws,
                              size_t ws_size, hipStream_t stream) {
  (void)in_sizes;
  (void)n_in;
  (void)out_size;
  (void)ws_size;
  const float* x = (const float*)d_in[0];
  const float* Wq = (const float*)d_in[1];
  const float* bq = (const float*)d_in[2];
  const float* Wk = (const float*)d_in[3];
  const float* bk = (const float*)d_in[4];
  const float* Wv = (const float*)d_in[5];
  const float* bv = (const float*)d_in[6];
  const float* Wo = (const float*)d_in[7];
  const float* bo = (const float*)d_in[8];
  const float* slopes = (const float*)d_in[9];
  float* out = (float*)d_out;

  // Workspace partition (bf16 elems): total 24M elems = 48 MB
  constexpr size_t NX = 4096ull * 1024;         // x bf16
  constexpr size_t NW3 = 3072ull * 1024;        // WqkvT bf16
  constexpr size_t NW1 = 1024ull * 1024;        // WoT bf16
  constexpr size_t NQ = 2ull * 16 * 2048 * 64;  // per Q/K/V buffer
  __bf16* xb = (__bf16*)d_ws;
  __bf16* wt = xb + NX;
  __bf16* wot = wt + NW3;
  __bf16* qb = wot + NW1;
  __bf16* kb = qb + NQ;
  __bf16* vb = kb + NQ;
  __bf16* obuf = vb + NQ;

  k_cvt<<<4096, 256, 0, stream>>>(x, xb);
  dim3 tg(32, 32);
  k_transpose<<<tg, 256, 0, stream>>>(Wq, wt);
  k_transpose<<<tg, 256, 0, stream>>>(Wk, wt + 1024ull * 1024);
  k_transpose<<<tg, 256, 0, stream>>>(Wv, wt + 2ull * 1024 * 1024);
  k_transpose<<<tg, 256, 0, stream>>>(Wo, wot);

  k_gemm_qkv<<<dim3(24, 32), 256, 0, stream>>>(xb, wt, bq, bk, bv, qb, kb, vb);
  k_flash<<<512, 256, 0, stream>>>(qb, kb, vb, slopes, obuf);
  k_gemm_out<<<dim3(8, 32), 256, 0, stream>>>(obuf, wot, bo, out);
}
